// Net_25864293057294
// MI455X (gfx1250) — matrix operand order repaired and checked by reading, not compiled, not run on hardware
//
#include <hip/hip_runtime.h>

// ---------------- problem constants ----------------
#define NN     100000
#define EE     1600000
#define TOT_E  (EE + NN)      // edges + self loops
#define F_IN   512
#define F_HID  64
#define F_OUT  40
#define NEG_SLOPE 0.2f

// ---------------- WMMA vector types ----------------
typedef __attribute__((ext_vector_type(16))) __bf16 v16bf;
typedef __attribute__((ext_vector_type(8)))  float  v8f;

// ---------------- bf16 helpers (RNE split for bf16x3 compensation) ------
static __device__ __forceinline__ __bf16 f2bf(float f) {
    union { float f; unsigned u; } v; v.f = f;
    unsigned r = (v.u + 0x7FFFu + ((v.u >> 16) & 1u)) >> 16;
    union { unsigned short s; __bf16 b; } o; o.s = (unsigned short)r;
    return o.b;
}
static __device__ __forceinline__ float bf2f(__bf16 b) {
    union { unsigned short s; __bf16 b; } i; i.b = b;
    union { unsigned u; float f; } o; o.u = ((unsigned)i.s) << 16;
    return o.f;
}

// order-preserving float<->uint encode for atomicMax on signed floats
static __device__ __forceinline__ unsigned encf(float f) {
    unsigned u = __float_as_uint(f);
    return (u & 0x80000000u) ? ~u : (u | 0x80000000u);
}
static __device__ __forceinline__ float decf(unsigned e) {
    unsigned u = (e & 0x80000000u) ? (e ^ 0x80000000u) : ~e;
    return __uint_as_float(u);
}
#define ENC_NEG_INF 0x007FFFFFu   // encf(-inf)

static __device__ __forceinline__ float leaky(float z) {
    return z > 0.f ? z : NEG_SLOPE * z;
}

// Build bf16 hi/lo WMMA A-operand from 4 float4 runs (contiguous A layout:
// element i -> K = 16*hf + i, i.e. one 16-wide contiguous run).
static __device__ __forceinline__ void split4(float4 g, int base,
                                              v16bf& hi, v16bf& lo) {
    float v[4] = {g.x, g.y, g.z, g.w};
    #pragma unroll
    for (int j = 0; j < 4; ++j) {
        __bf16 h = f2bf(v[j]);
        hi[base + j] = h;
        lo[base + j] = f2bf(v[j] - bf2f(h));
    }
}

// =====================================================================
// GEMM1: h1[N,64] = x[N,512] @ W1[512,64], bf16x3 compensated WMMA.
// Block = 256 threads = 8 waves; each wave owns a 16x64 C tile.
// B fragments pre-swizzled in LDS: frag[(kstep*4+nt)*32 + lane][16]
// so the per-lane operand is one contiguous 32B v16bf (-> ds_load_b128 x2).
// =====================================================================
#define KSLAB 128
#define NKS   (KSLAB / 32)
__global__ void gemm1_wmma(const float* __restrict__ x,
                           const float* __restrict__ W,
                           float* __restrict__ h) {
    __shared__ __align__(32) __bf16 BhiS[NKS * 4 * 32 * 16];
    __shared__ __align__(32) __bf16 BloS[NKS * 4 * 32 * 16];

    const int t    = threadIdx.x;
    const int wave = t >> 5;
    const int lane = t & 31;          // lane == hf*16 + c
    const int c    = lane & 15;
    const int hf   = lane >> 4;

    const int row0 = blockIdx.x * 128 + wave * 16;
    int row = row0 + c; if (row >= NN) row = NN - 1;   // clamp loads, guard stores
    const float* xr = x + (size_t)row * F_IN;

    v8f acc[4] = {};

    for (int kk = 0; kk < F_IN; kk += KSLAB) {
        __syncthreads();
        // stage W slab, swizzled into per-lane WMMA B fragments
        for (int i = t; i < KSLAB * F_HID; i += 256) {
            int k = i >> 6, n = i & 63;
            float w = W[(kk + k) * F_HID + n];
            int kstep = k >> 5, krem = k & 31;
            int flane = ((krem >> 4) << 4) | (n & 15);   // hf*16 + c
            int idx = ((((kstep << 2) | (n >> 4)) << 5) | flane) * 16 + (krem & 15);
            __bf16 hi = f2bf(w);
            BhiS[idx] = hi;
            BloS[idx] = f2bf(w - bf2f(hi));
        }
        __syncthreads();

        #pragma unroll
        for (int ks = 0; ks < NKS; ++ks) {
            const int k0 = kk + ks * 32;
            // A operand: one 16-wide contiguous K run per lane (K = 16*hf + i) -> 4x float4
            const float4* xv0 = (const float4*)(xr + k0 + hf * 16);
            const float4* xv1 = (const float4*)(xr + k0 + hf * 16 + 8);
            float4 a0 = xv0[0], a1 = xv0[1], a2 = xv1[0], a3 = xv1[1];
            v16bf Ahi, Alo;
            split4(a0, 0, Ahi, Alo);
            split4(a1, 4, Ahi, Alo);
            split4(a2, 8, Ahi, Alo);
            split4(a3, 12, Ahi, Alo);

            #pragma unroll
            for (int nt = 0; nt < 4; ++nt) {
                const int fidx = ((((ks << 2) | nt) << 5) | lane) * 16;
                v16bf Bhi = *(const v16bf*)&BhiS[fidx];
                v16bf Blo = *(const v16bf*)&BloS[fidx];
                acc[nt] = __builtin_amdgcn_wmma_f32_16x16x32_bf16(
                    false, Alo, false, Bhi, (short)0, acc[nt], false, false);
                acc[nt] = __builtin_amdgcn_wmma_f32_16x16x32_bf16(
                    false, Ahi, false, Blo, (short)0, acc[nt], false, false);
                acc[nt] = __builtin_amdgcn_wmma_f32_16x16x32_bf16(
                    false, Ahi, false, Bhi, (short)0, acc[nt], false, false);
            }
        }
    }
    // C/D layout: VGPR r holds M = r + 8*hf, N = c
    #pragma unroll
    for (int nt = 0; nt < 4; ++nt)
        #pragma unroll
        for (int r = 0; r < 8; ++r) {
            int rr = row0 + r + 8 * hf;
            if (rr < NN) h[(size_t)rr * F_HID + nt * 16 + c] = acc[nt][r];
        }
}

// =====================================================================
// GEMM2: h2[N,40] = hin[N,64] @ W2[64,40] (N padded to 48), same scheme.
// frag[(kstep*3+nt)*32 + lane][16], kstep in {0,1}, nt in {0,1,2}
// =====================================================================
__global__ void gemm2_wmma(const float* __restrict__ hin,
                           const float* __restrict__ W,
                           float* __restrict__ h2) {
    __shared__ __align__(32) __bf16 BhiS[2 * 3 * 32 * 16];
    __shared__ __align__(32) __bf16 BloS[2 * 3 * 32 * 16];

    const int t = threadIdx.x;
    for (int i = t; i < F_HID * 48; i += 256) {
        int k = i / 48, n = i - (i / 48) * 48;
        float w = (n < F_OUT) ? W[k * F_OUT + n] : 0.f;
        int kstep = k >> 5, krem = k & 31;
        int flane = ((krem >> 4) << 4) | (n & 15);
        int idx = (((kstep * 3 + (n >> 4)) << 5) | flane) * 16 + (krem & 15);
        __bf16 hi = f2bf(w);
        BhiS[idx] = hi;
        BloS[idx] = f2bf(w - bf2f(hi));
    }
    __syncthreads();

    const int wave = t >> 5, lane = t & 31;
    const int c = lane & 15, hf = lane >> 4;
    const int row0 = blockIdx.x * 128 + wave * 16;
    int row = row0 + c; if (row >= NN) row = NN - 1;
    const float* ar = hin + (size_t)row * F_HID;

    v8f acc[3] = {};
    #pragma unroll
    for (int ks = 0; ks < 2; ++ks) {
        const int k0 = ks * 32;
        const float4* xv0 = (const float4*)(ar + k0 + hf * 16);
        const float4* xv1 = (const float4*)(ar + k0 + hf * 16 + 8);
        float4 a0 = xv0[0], a1 = xv0[1], a2 = xv1[0], a3 = xv1[1];
        v16bf Ahi, Alo;
        split4(a0, 0, Ahi, Alo);
        split4(a1, 4, Ahi, Alo);
        split4(a2, 8, Ahi, Alo);
        split4(a3, 12, Ahi, Alo);

        #pragma unroll
        for (int nt = 0; nt < 3; ++nt) {
            const int fidx = (((ks * 3 + nt) << 5) | lane) * 16;
            v16bf Bhi = *(const v16bf*)&BhiS[fidx];
            v16bf Blo = *(const v16bf*)&BloS[fidx];
            acc[nt] = __builtin_amdgcn_wmma_f32_16x16x32_bf16(
                false, Alo, false, Bhi, (short)0, acc[nt], false, false);
            acc[nt] = __builtin_amdgcn_wmma_f32_16x16x32_bf16(
                false, Ahi, false, Blo, (short)0, acc[nt], false, false);
            acc[nt] = __builtin_amdgcn_wmma_f32_16x16x32_bf16(
                false, Ahi, false, Bhi, (short)0, acc[nt], false, false);
        }
    }
    #pragma unroll
    for (int nt = 0; nt < 3; ++nt)
        #pragma unroll
        for (int r = 0; r < 8; ++r) {
            int rr = row0 + r + 8 * hf, n = nt * 16 + c;
            if (rr < NN && n < F_OUT) h2[(size_t)rr * F_OUT + n] = acc[nt][r];
        }
}

// ---------------- attention projections: one wave per node -----------
template<int F>
__global__ void alpha_proj(const float* __restrict__ h,
                           const float* __restrict__ a_src,
                           const float* __restrict__ a_dst,
                           float* __restrict__ as, float* __restrict__ ad) {
    int node = blockIdx.x * 8 + (threadIdx.x >> 5);
    int lane = threadIdx.x & 31;
    if (node >= NN) return;
    float s1 = 0.f, s2 = 0.f;
    for (int f = lane; f < F; f += 32) {
        float v = h[(size_t)node * F + f];
        s1 += v * a_src[f];
        s2 += v * a_dst[f];
    }
    #pragma unroll
    for (int o = 16; o; o >>= 1) {
        s1 += __shfl_xor(s1, o, 32);
        s2 += __shfl_xor(s2, o, 32);
    }
    if (lane == 0) { as[node] = s1; ad[node] = s2; }
}

// ---------------- init kernels (ws is poisoned by harness) -----------
__global__ void init_ms(unsigned* __restrict__ m, float* __restrict__ s) {
    int i = blockIdx.x * 256 + threadIdx.x;
    if (i < NN) { m[i] = ENC_NEG_INF; s[i] = 0.f; }
}
__global__ void zero_f(float* __restrict__ p, int n) {
    int i = blockIdx.x * 256 + threadIdx.x;
    if (i < n) p[i] = 0.f;
}

// ---------------- edge phase ----------------
// one-shot: int64 edge_index -> int32 src/dst (+ synthesized self loops)
__global__ void edge_prep_k(const long long* __restrict__ ei,
                            int* __restrict__ src32, int* __restrict__ dst32) {
    int e = blockIdx.x * 256 + threadIdx.x;
    if (e >= TOT_E) return;
    if (e < EE) { src32[e] = (int)ei[e]; dst32[e] = (int)ei[EE + e]; }
    else        { src32[e] = dst32[e] = e - EE; }
}

__global__ void edge_max_k(const float* __restrict__ as, const float* __restrict__ ad,
                           const int* __restrict__ src32, const int* __restrict__ dst32,
                           unsigned* __restrict__ m) {
    int e = blockIdx.x * 256 + threadIdx.x;
    if (e >= TOT_E) return;
    int dst = dst32[e];
    float z = leaky(as[src32[e]] + ad[dst]);
    atomicMax(&m[dst], encf(z));
}

__global__ void edge_weight_k(const float* __restrict__ as, const float* __restrict__ ad,
                              const int* __restrict__ src32, const int* __restrict__ dst32,
                              const unsigned* __restrict__ m,
                              float* __restrict__ w, float* __restrict__ s) {
    int e = blockIdx.x * 256 + threadIdx.x;
    if (e >= TOT_E) return;
    int dst = dst32[e];
    float z  = leaky(as[src32[e]] + ad[dst]);
    float ww = expf(z - decf(m[dst]));
    w[e] = ww;
    atomicAdd(&s[dst], ww);
}

template<int F>
__global__ void edge_scatter_k(const float* __restrict__ w,
                               const int* __restrict__ src32, const int* __restrict__ dst32,
                               const float* __restrict__ h, float* __restrict__ agg) {
    int tid = blockIdx.x * 256 + threadIdx.x;
    if (tid >= TOT_E * F) return;
    int e = tid / F, f = tid - e * F;       // consecutive threads -> consecutive f
    atomicAdd(&agg[(size_t)dst32[e] * F + f], w[e] * h[(size_t)src32[e] * F + f]);
}

// layer-1 finalize: agg/s + b, relu (in place)
__global__ void finalize1_k(float* __restrict__ agg, const float* __restrict__ s,
                            const float* __restrict__ b) {
    int tid = blockIdx.x * 256 + threadIdx.x;
    if (tid >= NN * F_HID) return;
    int node = tid >> 6, f = tid & 63;
    float v = agg[tid] / s[node] + b[f];
    agg[tid] = v > 0.f ? v : 0.f;
}

// layer-2 finalize + log_softmax: one wave per node (40 classes)
__global__ void finalize2_k(const float* __restrict__ agg, const float* __restrict__ s,
                            const float* __restrict__ b, float* __restrict__ out) {
    int node = blockIdx.x * 8 + (threadIdx.x >> 5);
    int lane = threadIdx.x & 31;
    if (node >= NN) return;
    float inv = 1.f / s[node];
    float z0 = -INFINITY, z1 = -INFINITY;
    if (lane < F_OUT)      z0 = agg[(size_t)node * F_OUT + lane]      * inv + b[lane];
    if (lane + 32 < F_OUT) z1 = agg[(size_t)node * F_OUT + lane + 32] * inv + b[lane + 32];
    float mx = fmaxf(z0, z1);
    #pragma unroll
    for (int o = 16; o; o >>= 1) mx = fmaxf(mx, __shfl_xor(mx, o, 32));
    float e0 = (lane < F_OUT)      ? expf(z0 - mx) : 0.f;
    float e1 = (lane + 32 < F_OUT) ? expf(z1 - mx) : 0.f;
    float se = e0 + e1;
    #pragma unroll
    for (int o = 16; o; o >>= 1) se += __shfl_xor(se, o, 32);
    float lse = logf(se) + mx;
    if (lane < F_OUT)      out[(size_t)node * F_OUT + lane]      = z0 - lse;
    if (lane + 32 < F_OUT) out[(size_t)node * F_OUT + lane + 32] = z1 - lse;
}

// =====================================================================
extern "C" void kernel_launch(void* const* d_in, const int* in_sizes, int n_in,
                              void* d_out, int out_size, void* d_ws, size_t ws_size,
                              hipStream_t stream) {
    const float*     x    = (const float*)d_in[0];
    const long long* ei   = (const long long*)d_in[1];   // int64 edge_index [2,E]
    const float*     W1   = (const float*)d_in[2];
    const float*     pas1 = (const float*)d_in[3];
    const float*     pad1 = (const float*)d_in[4];
    const float*     b1   = (const float*)d_in[5];
    const float*     W2   = (const float*)d_in[6];
    const float*     pas2 = (const float*)d_in[7];
    const float*     pad2 = (const float*)d_in[8];
    const float*     b2   = (const float*)d_in[9];
    float*           out  = (float*)d_out;

    // workspace carve-up (floats/ints): ~105 MB total
    float* h1    = (float*)d_ws;                    // N*64
    float* agg1  = h1    + (size_t)NN * F_HID;      // N*64 (relu'd layer-2 input)
    float* h2    = agg1  + (size_t)NN * F_HID;      // N*40
    float* agg2  = h2    + (size_t)NN * F_OUT;      // N*40
    float* as    = agg2  + (size_t)NN * F_OUT;      // N
    float* ad    = as    + NN;                      // N
    unsigned* m  = (unsigned*)(ad + NN);            // N
    float* sden  = (float*)(m + NN);                // N
    float* w_e   = sden  + NN;                      // E+N
    int*   src32 = (int*)(w_e + TOT_E);             // E+N
    int*   dst32 = src32 + TOT_E;                   // E+N

    const int GEMM_GRID  = (NN + 127) / 128;
    const int NODE_WAVES = (NN + 7) / 8;
    const int NODE_GRID  = (NN + 255) / 256;
    const int EDGE_GRID  = (TOT_E + 255) / 256;

    edge_prep_k<<<EDGE_GRID, 256, 0, stream>>>(ei, src32, dst32);

    // ---------------- Layer 1 ----------------
    gemm1_wmma<<<GEMM_GRID, 256, 0, stream>>>(x, W1, h1);
    alpha_proj<F_HID><<<NODE_WAVES, 256, 0, stream>>>(h1, pas1, pad1, as, ad);
    init_ms<<<NODE_GRID, 256, 0, stream>>>(m, sden);
    zero_f<<<(NN * F_HID + 255) / 256, 256, 0, stream>>>(agg1, NN * F_HID);
    edge_max_k<<<EDGE_GRID, 256, 0, stream>>>(as, ad, src32, dst32, m);
    edge_weight_k<<<EDGE_GRID, 256, 0, stream>>>(as, ad, src32, dst32, m, w_e, sden);
    edge_scatter_k<F_HID><<<(TOT_E * F_HID + 255) / 256, 256, 0, stream>>>(w_e, src32, dst32, h1, agg1);
    finalize1_k<<<(NN * F_HID + 255) / 256, 256, 0, stream>>>(agg1, sden, b1);

    // ---------------- Layer 2 ----------------
    gemm2_wmma<<<GEMM_GRID, 256, 0, stream>>>(agg1, W2, h2);
    alpha_proj<F_OUT><<<NODE_WAVES, 256, 0, stream>>>(h2, pas2, pad2, as, ad);
    init_ms<<<NODE_GRID, 256, 0, stream>>>(m, sden);
    zero_f<<<(NN * F_OUT + 255) / 256, 256, 0, stream>>>(agg2, NN * F_OUT);
    edge_max_k<<<EDGE_GRID, 256, 0, stream>>>(as, ad, src32, dst32, m);
    edge_weight_k<<<EDGE_GRID, 256, 0, stream>>>(as, ad, src32, dst32, m, w_e, sden);
    edge_scatter_k<F_OUT><<<(TOT_E * F_OUT + 255) / 256, 256, 0, stream>>>(w_e, src32, dst32, h2, agg2);
    finalize2_k<<<NODE_WAVES, 256, 0, stream>>>(agg2, sden, b2, out);
}
